// MultiHeadAttentioWithRPE_64166811402559
// MI455X (gfx1250) — matrix operand order repaired and checked by reading, not compiled, not run on hardware
//
#include <hip/hip_runtime.h>

// ---------------------------------------------------------------------------
// MI455X (gfx1250) implementation of MHA with relative positional encoding.
// bf16 WMMA (v_wmma_f32_16x16x32_bf16) for all GEMMs + flash attention.
// All B operands pre-transposed so every fragment load is a 32-bit pair load
// with uniform-base + 32-bit offset addressing (saddr form).
// ---------------------------------------------------------------------------

typedef unsigned short u16;
typedef unsigned int   u32;
typedef __attribute__((ext_vector_type(16))) __bf16 v16bf;
typedef __attribute__((ext_vector_type(8)))  float  v8f;

union FragBF { u32 u[8]; v16bf v; };

#define SEQ     2048
#define BATCH   2
#define DMODEL  1024
#define NHEAD   8
#define FEAT    128

__device__ __forceinline__ u16 f2bf(float f) {
    u32 u = __float_as_uint(f);
    u32 rnd = 0x7FFFu + ((u >> 16) & 1u);   // round-to-nearest-even
    return (u16)((u + rnd) >> 16);
}

// --- WMMA fragment loaders (layouts per CDNA5 ISA 7.12.2, wave32) -----------
// All offsets are 32-bit element indices from a uniform base pointer so the
// backend can use scalar-base + voffset global loads.
// A (16x32, bf16): lane m = lane&15, half = lane>>4;
//   reg p holds K = (p>>2)*16 + half*8 + (p&3)*2  (consecutive K pair)
__device__ __forceinline__ void load_frag_a(const u16* base, int ld, int row0,
                                            int k0, u32* r, int lane) {
    int m = lane & 15, half = lane >> 4;
    int off = (row0 + m) * ld + k0 + (half << 3);
#pragma unroll
    for (int p = 0; p < 8; ++p) {
        int k = ((p >> 2) << 4) + ((p & 3) << 1);
        r[p] = *(const u32*)(base + off + k);
    }
}

// B (32x16, bf16) from transposed row-major [N,K] storage:
//   lane n = lane&15, half = lane>>4; reg p holds K = (p>>2)*16 + half*8 + (p&3)*2, +1
__device__ __forceinline__ void load_frag_b_t(const u16* base, int ld, int n0,
                                              int k0, u32* r, int lane) {
    int n = lane & 15, half = lane >> 4;
    int off = (n0 + n) * ld + k0 + (half << 3);
#pragma unroll
    for (int p = 0; p < 8; ++p)
        r[p] = *(const u32*)(base + off + (((p >> 2) << 4) + ((p & 3) << 1)));
}

// --- fp32 -> bf16 with transpose: dst[n*K + k] = src[k*N + n] ---------------
__global__ void f2bfT_kernel(const float* __restrict__ src, u16* __restrict__ dst,
                             int K, int N) {
    int idx = blockIdx.x * blockDim.x + threadIdx.x;
    if (idx < K * N) {
        int n = idx / K, k = idx - n * K;
        dst[idx] = f2bf(src[(size_t)k * N + n]);
    }
}

// --- LayerNorm (eps=1e-5), fp32 in -> bf16 out ------------------------------
__global__ __launch_bounds__(256) void ln_kernel(const float* __restrict__ x,
                                                 const float* __restrict__ g,
                                                 const float* __restrict__ be,
                                                 u16* __restrict__ h) {
    const int D = DMODEL;
    int row = blockIdx.x;
    const float* xr = x + (size_t)row * D;
    float s = 0.f, s2 = 0.f;
    for (int d = threadIdx.x; d < D; d += 256) {
        float t = xr[d];
        s += t; s2 += t * t;
    }
    __shared__ float sh1[256], sh2[256];
    sh1[threadIdx.x] = s; sh2[threadIdx.x] = s2;
    __syncthreads();
    for (int off = 128; off > 0; off >>= 1) {
        if ((int)threadIdx.x < off) {
            sh1[threadIdx.x] += sh1[threadIdx.x + off];
            sh2[threadIdx.x] += sh2[threadIdx.x + off];
        }
        __syncthreads();
    }
    float mu  = sh1[0] * (1.f / D);
    float var = sh2[0] * (1.f / D) - mu * mu;
    float rs  = rsqrtf(var + 1e-5f);
    u16* hr = h + (size_t)row * D;
    for (int d = threadIdx.x; d < D; d += 256)
        hr[d] = f2bf((xr[d] - mu) * rs * g[d] + be[d]);
}

// --- sinusoid PE, flipped along position, bf16 ------------------------------
__global__ __launch_bounds__(256) void pe_kernel(u16* __restrict__ pe) {
    const int D = DMODEL;
    int i = blockIdx.x;
    float p = (float)(SEQ - 1 - i);   // pe[::-1]
    for (int d = threadIdx.x; d < D; d += 256) {
        int m2 = d & ~1;              // 2*(d>>1)
        float div = __expf((float)m2 * (-9.210340371976184f / (float)D));
        float ang = p * div;
        float val = (d & 1) ? __cosf(ang) : __sinf(ang);
        pe[(size_t)i * D + d] = f2bf(val);
    }
}

// --- generic bf16 GEMM: C[M,N] = A[M,K] * Bt[N,K]^T -------------------------
// block = 128 threads (4 waves); wave computes 32 rows x 64 cols.
// grid.x = N/64, grid.y = M/128.
// mode: 0 bf16 store; 1 q/qu/qv projection; 2 fp32+bias (final out);
//       3 fp32 plain (pos); 4 bf16 store transposed per (b,col) for V.
__global__ __launch_bounds__(128) void gemm_bf16_kernel(
    const u16* __restrict__ A, int lda,
    const u16* __restrict__ Bt, int ldb,
    int K, int mode,
    u16* __restrict__ Cb, float* __restrict__ Cf, int ldc,
    const float* __restrict__ bias,
    u16* __restrict__ quo, u16* __restrict__ qvo,
    const float* __restrict__ ub, const float* __restrict__ vb) {
    int lane = threadIdx.x & 31;
    int wave = threadIdx.x >> 5;
    int m0 = blockIdx.y * 128 + wave * 32;
    int n0 = blockIdx.x * 64;
    int half = lane >> 4, nIdx = lane & 15;

    v8f zero = {0.f, 0.f, 0.f, 0.f, 0.f, 0.f, 0.f, 0.f};
    v8f acc[2][4];
#pragma unroll
    for (int mt = 0; mt < 2; ++mt)
#pragma unroll
        for (int nt = 0; nt < 4; ++nt) acc[mt][nt] = zero;

    for (int kk = 0; kk < K; kk += 32) {
        FragBF a0, a1;
        load_frag_a(A, lda, m0,      kk, a0.u, lane);
        load_frag_a(A, lda, m0 + 16, kk, a1.u, lane);
#pragma unroll
        for (int nt = 0; nt < 4; ++nt) {
            FragBF bf;
            load_frag_b_t(Bt, ldb, n0 + nt * 16, kk, bf.u, lane);
            acc[0][nt] = __builtin_amdgcn_wmma_f32_16x16x32_bf16(
                false, a0.v, false, bf.v, (short)0, acc[0][nt], false, false);
            acc[1][nt] = __builtin_amdgcn_wmma_f32_16x16x32_bf16(
                false, a1.v, false, bf.v, (short)0, acc[1][nt], false, false);
        }
    }

#pragma unroll
    for (int mt = 0; mt < 2; ++mt) {
#pragma unroll
        for (int nt = 0; nt < 4; ++nt) {
#pragma unroll
            for (int p = 0; p < 8; ++p) {
                int row = m0 + mt * 16 + half * 8 + p;
                int col = n0 + nt * 16 + nIdx;
                float val = acc[mt][nt][p];
                if (mode == 0) {
                    Cb[(size_t)row * ldc + col] = f2bf(val);
                } else if (mode == 1) {
                    Cb[(size_t)row * ldc + col]  = f2bf(val);            // q (=k)
                    quo[(size_t)row * ldc + col] = f2bf(val + ub[col]);  // q+u
                    if (col < 2 * FEAT)                                  // heads 0,1
                        qvo[(size_t)row * (2 * FEAT) + col] = f2bf(val + vb[col]);
                } else if (mode == 2) {
                    Cf[(size_t)row * ldc + col] = val + bias[col];
                } else if (mode == 3) {
                    Cf[(size_t)row * ldc + col] = val;
                } else {
                    // V transposed: row = i*BATCH + b, col = h*F + f
                    // vT[(b*DMODEL + col)*SEQ + i]
                    int b = row & 1, i = row >> 1;
                    Cb[((size_t)(b * DMODEL + col)) * SEQ + i] = f2bf(val);
                }
            }
        }
    }
}

// --- relative-shift pos lookup (derived closed form of _left_shift) ---------
__device__ __forceinline__ float pos_lookup(const float* __restrict__ posc,
                                            int i, int j) {
    if (j <= i)     return posc[i * SEQ + (SEQ - 1 - i + j)];
    if (j == i + 1) return 0.f;
    return posc[(i + 1) * SEQ + (j - i - 2)];
}

// --- flash attention: one wave per (16-row i-block, b, h) -------------------
__global__ __launch_bounds__(128) void attn_kernel(
    const u16* __restrict__ quB,  // q + u_bias (bf16) [I,B,H*F]
    const u16* __restrict__ qB,   // q (= k)    (bf16) [I,B,H*F]
    const u16* __restrict__ vT,   // v transposed (bf16) [B,H*F,SEQ]
    const float* __restrict__ pos,// [3][SEQ][SEQ] for (b,n)=(0,0),(1,0),(1,1)
    u16* __restrict__ oB) {       // out (bf16) [I,B,H*F]
    __shared__ u16 pbuf[4][16 * 32];    // per-wave P tile (16 x 32 bf16)

    int lane = threadIdx.x & 31;
    int wave = threadIdx.x >> 5;
    int task = blockIdx.x * 4 + wave;   // 2048 tasks
    int ib = task >> 4;
    int bh = task & 15;
    int b = bh >> 3, h = bh & 7;
    int i0 = ib * 16;
    int half = lane >> 4, nIdx = lane & 15;
    const int ld = BATCH * DMODEL;      // row stride in i

    const u16* quBase = quB + b * DMODEL + h * FEAT;
    const u16* kBase  = qB + b * DMODEL + h * FEAT;
    const u16* vBase  = vT + (size_t)(b * DMODEL + h * FEAT) * SEQ;
    const float* posc = nullptr;        // tril over (B,N): keep n <= b
    if (h <= b) posc = pos + (size_t)((b == 0) ? 0 : (h == 0 ? 1 : 2)) * SEQ * SEQ;

    // preload (q+u) A-fragments covering F=128 (4 chunks of K=32)
    FragBF qa[4];
#pragma unroll
    for (int t = 0; t < 4; ++t)
        load_frag_a(quBase, ld, i0, t * 32, qa[t].u, lane);

    v8f zero = {0.f, 0.f, 0.f, 0.f, 0.f, 0.f, 0.f, 0.f};
    v8f acc[8];
#pragma unroll
    for (int ft = 0; ft < 8; ++ft) acc[ft] = zero;
    float mrow[8], lrow[8];
#pragma unroll
    for (int p = 0; p < 8; ++p) { mrow[p] = -3.0e38f; lrow[p] = 0.f; }

    const float scale = 0.03125f;       // 1/sqrt(1024)

    for (int jb = 0; jb < SEQ / 32; ++jb) {
        int j0 = jb * 32;
        v8f s0 = zero, s1 = zero;
#pragma unroll
        for (int t = 0; t < 4; ++t) {   // S = (q+u) * k^T
            FragBF k0f, k1f;
            load_frag_b_t(kBase, ld, j0,      t * 32, k0f.u, lane);
            load_frag_b_t(kBase, ld, j0 + 16, t * 32, k1f.u, lane);
            s0 = __builtin_amdgcn_wmma_f32_16x16x32_bf16(
                false, qa[t].v, false, k0f.v, (short)0, s0, false, false);
            s1 = __builtin_amdgcn_wmma_f32_16x16x32_bf16(
                false, qa[t].v, false, k1f.v, (short)0, s1, false, false);
        }
#pragma unroll
        for (int p = 0; p < 8; ++p) {   // online softmax over j
            int i = i0 + half * 8 + p;
            float e0 = s0[p], e1 = s1[p];
            if (posc) {
                int j = j0 + nIdx;
                e0 += pos_lookup(posc, i, j);
                e1 += pos_lookup(posc, i, j + 16);
            }
            e0 *= scale; e1 *= scale;
            float tm = fmaxf(e0, e1);
#pragma unroll
            for (int m = 8; m >= 1; m >>= 1)        // reduce within 16-lane half
                tm = fmaxf(tm, __shfl_xor(tm, m, 32));
            float newm = fmaxf(mrow[p], tm);
            float alpha = __expf(mrow[p] - newm);
            float p0 = __expf(e0 - newm);
            float p1 = __expf(e1 - newm);
            float rs = p0 + p1;
#pragma unroll
            for (int m = 8; m >= 1; m >>= 1)
                rs += __shfl_xor(rs, m, 32);
            lrow[p] = lrow[p] * alpha + rs;
            mrow[p] = newm;
#pragma unroll
            for (int ft = 0; ft < 8; ++ft) acc[ft][p] *= alpha;
            int mloc = half * 8 + p;
            pbuf[wave][mloc * 32 + nIdx]      = f2bf(p0);
            pbuf[wave][mloc * 32 + nIdx + 16] = f2bf(p1);
        }
        // per-wave LDS is in-order; wait so cross-lane reads see the stores
        asm volatile("s_wait_dscnt 0" ::: "memory");
        FragBF pa;
        load_frag_a(&pbuf[wave][0], 32, 0, 0, pa.u, lane);
#pragma unroll
        for (int ft = 0; ft < 8; ++ft) {            // O += P * V
            FragBF vf;
            load_frag_b_t(vBase, SEQ, ft * 16, j0, vf.u, lane);
            acc[ft] = __builtin_amdgcn_wmma_f32_16x16x32_bf16(
                false, pa.v, false, vf.v, (short)0, acc[ft], false, false);
        }
    }

#pragma unroll
    for (int ft = 0; ft < 8; ++ft) {
#pragma unroll
        for (int p = 0; p < 8; ++p) {
            int i = i0 + half * 8 + p;
            float val = acc[ft][p] / lrow[p];
            oB[((size_t)i * BATCH + b) * DMODEL + h * FEAT + ft * 16 + nIdx] =
                f2bf(val);
        }
    }
}

// ---------------------------------------------------------------------------
extern "C" void kernel_launch(void* const* d_in, const int* in_sizes, int n_in,
                              void* d_out, int out_size, void* d_ws,
                              size_t ws_size, hipStream_t stream) {
    const float* x     = (const float*)d_in[0];
    const float* gamma = (const float*)d_in[1];
    const float* beta  = (const float*)d_in[2];
    const float* Wq    = (const float*)d_in[3];
    const float* Wv    = (const float*)d_in[4];
    const float* Wr    = (const float*)d_in[5];
    const float* ubias = (const float*)d_in[6];
    const float* vbias = (const float*)d_in[7];
    const float* Wout  = (const float*)d_in[8];
    const float* bout  = (const float*)d_in[9];
    float* out = (float*)d_out;

    const int M = SEQ * BATCH;        // 4096 flattened (i,b) rows
    const int D = DMODEL;             // 1024
    char* ws = (char*)d_ws;
    auto take = [&](size_t bytes) {
        char* p = ws;
        ws += (bytes + 255) & ~(size_t)255;
        return p;
    };
    u16* WqT   = (u16*)take((size_t)D * D * 2);   // transposed [N,K] bf16
    u16* WvT   = (u16*)take((size_t)D * D * 2);
    u16* WrT   = (u16*)take((size_t)D * D * 2);
    u16* WoutT = (u16*)take((size_t)D * D * 2);
    u16* hB    = (u16*)take((size_t)M * D * 2);
    u16* peB   = (u16*)take((size_t)SEQ * D * 2);
    u16* qB    = (u16*)take((size_t)M * D * 2);
    u16* quB   = (u16*)take((size_t)M * D * 2);
    u16* qvB   = (u16*)take((size_t)M * 2 * FEAT * 2);     // heads 0,1 only
    u16* vTB   = (u16*)take((size_t)M * D * 2);            // [B,H*F,SEQ]
    u16* rB    = (u16*)take((size_t)SEQ * 2 * FEAT * 2);   // heads 0,1 only
    u16* oB    = (u16*)take((size_t)M * D * 2);
    float* posW = (float*)take((size_t)3 * SEQ * SEQ * 4); // 48 MB

    // 1. weights -> transposed bf16
    f2bfT_kernel<<<(D * D) / 256, 256, 0, stream>>>(Wq, WqT, D, D);
    f2bfT_kernel<<<(D * D) / 256, 256, 0, stream>>>(Wv, WvT, D, D);
    f2bfT_kernel<<<(D * D) / 256, 256, 0, stream>>>(Wr, WrT, D, D);
    f2bfT_kernel<<<(D * D) / 256, 256, 0, stream>>>(Wout, WoutT, D, D);
    // 2. LayerNorm + positional encoding
    ln_kernel<<<M, 256, 0, stream>>>(x, gamma, beta, hB);
    pe_kernel<<<SEQ, 256, 0, stream>>>(peB);
    // 3. projections: q (=k), q+u, q+v (heads 0,1);  v (stored transposed)
    gemm_bf16_kernel<<<dim3(D / 64, M / 128), 128, 0, stream>>>(
        hB, D, WqT, D, D, 1, qB, nullptr, D, nullptr, quB, qvB, ubias, vbias);
    gemm_bf16_kernel<<<dim3(D / 64, M / 128), 128, 0, stream>>>(
        hB, D, WvT, D, D, 4, vTB, nullptr, D, nullptr, nullptr, nullptr,
        nullptr, nullptr);
    // 4. r = pe @ Wr[:, 0:256]   (only heads 0,1 survive the (B,N) tril)
    gemm_bf16_kernel<<<dim3((2 * FEAT) / 64, SEQ / 128), 128, 0, stream>>>(
        peB, D, WrT, D, D, 0, rB, nullptr, 2 * FEAT, nullptr, nullptr,
        nullptr, nullptr, nullptr);
    // 5. pos matrices for (b,n) = (0,0), (1,0), (1,1):  (q+v) @ r^T
    for (int c = 0; c < 3; ++c) {
        int bb = (c == 0) ? 0 : 1;
        int nn = (c == 2) ? 1 : 0;
        gemm_bf16_kernel<<<dim3(SEQ / 64, SEQ / 128), 128, 0, stream>>>(
            qvB + bb * 2 * FEAT + nn * FEAT, 2 * 2 * FEAT,  // A: [i,f], lda=512
            rB + nn * FEAT, 2 * FEAT,                       // Bt: [jj,f], ldb=256
            FEAT, 3, nullptr, posW + (size_t)c * SEQ * SEQ, SEQ,
            nullptr, nullptr, nullptr, nullptr, nullptr);
    }
    // 6. flash attention (2048 wave-tasks, 4 waves/block)
    attn_kernel<<<(SEQ / 16) * BATCH * NHEAD / 4, 128, 0, stream>>>(
        quB, qB, vTB, posW, oB);
    // 7. out = o @ Wout + bout  (fp32)
    gemm_bf16_kernel<<<dim3(D / 64, M / 128), 128, 0, stream>>>(
        oB, D, WoutT, D, D, 2, nullptr, out, D, bout, nullptr, nullptr,
        nullptr, nullptr);
}
